// EnhancedGATCoverageDQN_43018392436914
// MI455X (gfx1250) — matrix operand order repaired and checked by reading, not compiled, not run on hardware
//
#include <hip/hip_runtime.h>
#include <hip/hip_bf16.h>

#define NNODES 10000
#define NEDGES 320000
#define DMODEL 256
#define NHEADS 4
#define CHEAD 64
#define NLAYERS 3
#define LN_EPS 1e-5f

typedef __attribute__((ext_vector_type(16))) __bf16 bf16x16;
typedef __attribute__((ext_vector_type(8)))  float  f32x8;

__device__ __forceinline__ unsigned short f2bf(float f) {
  unsigned u = __float_as_uint(f);
  unsigned r = u + 0x7FFFu + ((u >> 16) & 1u);
  return (unsigned short)(r >> 16);
}

// block-wide sum over 256 threads (all threads must participate)
__device__ __forceinline__ float block_sum_256(float v, float* sm) {
  int tid = threadIdx.x;
  sm[tid] = v;
  __syncthreads();
  #pragma unroll
  for (int off = 128; off > 0; off >>= 1) {
    if (tid < off) sm[tid] += sm[tid + off];
    __syncthreads();
  }
  float r = sm[0];
  __syncthreads();
  return r;
}

// ---------------------------------------------------------------------------
// Weight packing: gat_w [3][256][256] f32 -> bf16 WMMA B-fragment order.
// Fragment (kt, nt): lane l holds n = nt*16 + (l&15), k = kt*32 + (l>>4)*16 + e
// flat index: ((layer*128 + kt*16 + nt)*32 + lane)*16 + e  (contiguous 32B/lane)
// ---------------------------------------------------------------------------
__global__ void pack_gat_weights(const float* __restrict__ gat_w,
                                 unsigned short* __restrict__ wpack) {
  int idx = blockIdx.x * blockDim.x + threadIdx.x;
  if (idx >= NLAYERS * 65536) return;
  int layer = idx >> 16;
  int rem   = idx & 65535;
  int frag  = rem >> 9;          // kt*16 + nt
  int lane  = (rem >> 4) & 31;
  int e     = rem & 15;
  int kt = frag >> 4;
  int nt = frag & 15;
  int k = kt * 32 + (lane >> 4) * 16 + e;
  int n = nt * 16 + (lane & 15);
  wpack[idx] = f2bf(gat_w[layer * 65536 + k * 256 + n]);
}

// g[l][j][h] = sum_c edge_w[l][j][h*64+c] * att_edge[l][h][c]   (36 scalars)
__global__ void compute_edge_gain(const float* __restrict__ edge_w,
                                  const float* __restrict__ att_edge,
                                  float* __restrict__ g) {
  int t = threadIdx.x;
  if (t >= NLAYERS * 3 * NHEADS) return;
  int l = t / 12, r = t % 12, j = r / 4, hh = r % 4;
  float s = 0.f;
  for (int c = 0; c < CHEAD; ++c)
    s += edge_w[l * 3 * DMODEL + j * DMODEL + hh * CHEAD + c] *
         att_edge[l * DMODEL + hh * CHEAD + c];
  g[t] = s;
}

// ---------------------------------------------------------------------------
// Node encoder: h0 = relu(LN(x @ enc_w + enc_b)); also bf16 mirror + ctx sum
// ---------------------------------------------------------------------------
__global__ void encode_nodes(const float* __restrict__ x,
                             const float* __restrict__ enc_w,
                             const float* __restrict__ enc_b,
                             const float* __restrict__ enc_g,
                             const float* __restrict__ enc_bt,
                             float* __restrict__ h,
                             unsigned short* __restrict__ hbf,
                             float* __restrict__ ctx) {
  __shared__ float sm[256];
  __shared__ float xr[10];
  int row = blockIdx.x;
  int c = threadIdx.x;
  if (c < 10) xr[c] = x[row * 10 + c];
  __syncthreads();
  float acc = enc_b[c];
  #pragma unroll
  for (int k = 0; k < 10; ++k) acc += xr[k] * enc_w[k * 256 + c];
  float m = block_sum_256(acc, sm) * (1.f / 256.f);
  float d = acc - m;
  float var = block_sum_256(d * d, sm) * (1.f / 256.f);
  float v = d * rsqrtf(var + LN_EPS) * enc_g[c] + enc_bt[c];
  v = fmaxf(v, 0.f);
  h[row * 256 + c] = v;
  hbf[row * 256 + c] = f2bf(v);
  atomicAdd(&ctx[c], v);
}

// Adaptive virtual node MLP (single block)
__global__ void virtual_node(const float* __restrict__ ctx,
                             const float* __restrict__ vn_base,
                             const float* __restrict__ vn_w1,
                             const float* __restrict__ vn_b1,
                             const float* __restrict__ vn_g,
                             const float* __restrict__ vn_bt,
                             const float* __restrict__ vn_w2,
                             const float* __restrict__ vn_b2,
                             float* __restrict__ h,
                             unsigned short* __restrict__ hbf,
                             float* __restrict__ save_rows) {
  __shared__ float sm[256];
  __shared__ float s0[256];
  __shared__ float s1[256];
  int c = threadIdx.x;
  s0[c] = ctx[c] * (1.f / (float)NNODES);
  __syncthreads();
  float acc = vn_b1[c];
  for (int k = 0; k < 256; ++k) acc += s0[k] * vn_w1[k * 256 + c];
  float m = block_sum_256(acc, sm) * (1.f / 256.f);
  float d = acc - m;
  float var = block_sum_256(d * d, sm) * (1.f / 256.f);
  float t = d * rsqrtf(var + LN_EPS) * vn_g[c] + vn_bt[c];
  s1[c] = fmaxf(t, 0.f);
  __syncthreads();
  float acc2 = vn_b2[c];
  for (int k = 0; k < 256; ++k) acc2 += s1[k] * vn_w2[k * 256 + c];
  float vn = vn_base[c] + tanhf(acc2);
  h[NNODES * 256 + c] = vn;
  hbf[NNODES * 256 + c] = f2bf(vn);
  save_rows[c] = vn;   // outs[0] virtual-node row (jk only uses row N)
}

// ---------------------------------------------------------------------------
// Edge bookkeeping: degree + incoming-attr sum over E real + 2N virtual edges
// ---------------------------------------------------------------------------
__global__ void edge_degree(const int* __restrict__ ei,
                            const float* __restrict__ eattr,
                            float* __restrict__ deg,
                            float* __restrict__ asum) {
  int e = blockIdx.x * blockDim.x + threadIdx.x;
  const int M = NEDGES + 2 * NNODES;
  if (e >= M) return;
  int dst;
  if (e < NEDGES) {
    dst = ei[NEDGES + e];
    atomicAdd(&asum[dst * 3 + 0], eattr[e * 3 + 0]);
    atomicAdd(&asum[dst * 3 + 1], eattr[e * 3 + 1]);
    atomicAdd(&asum[dst * 3 + 2], eattr[e * 3 + 2]);
  } else if (e < NEDGES + NNODES) {
    dst = e - NEDGES;
    atomicAdd(&asum[dst * 3 + 0], 0.5f);
  } else {
    dst = NNODES;
    atomicAdd(&asum[dst * 3 + 0], 0.5f);
  }
  atomicAdd(&deg[dst], 1.f);
}

__global__ void finalize_loop_attr(const float* __restrict__ deg,
                                   const float* __restrict__ asum,
                                   float* __restrict__ lattr) {
  int n = blockIdx.x * blockDim.x + threadIdx.x;
  if (n >= NNODES + 1) return;
  float dv = 1.f / fmaxf(deg[n], 1.f);
  lattr[n * 3 + 0] = asum[n * 3 + 0] * dv;
  lattr[n * 3 + 1] = asum[n * 3 + 1] * dv;
  lattr[n * 3 + 2] = asum[n * 3 + 2] * dv;
}

// ---------------------------------------------------------------------------
// WMMA GEMM: xs[Nt,256] = hbf[Nt,256](bf16) @ wpack(layer)(bf16), f32 accum.
// Workgroup = 8 waves -> 32 rows x 256 cols; wave = 16 rows x 64 cols (4 tiles)
// ---------------------------------------------------------------------------
__global__ void gemm_xs_wmma(const unsigned short* __restrict__ hbf,
                             const unsigned short* __restrict__ wpack,
                             float* __restrict__ xs, int Nt) {
  int tid  = threadIdx.x;
  int wave = tid >> 5;
  int lane = tid & 31;
  int rt = wave >> 2;          // row half of block tile
  int cg = wave & 3;           // column quarter
  int rowbase = blockIdx.x * 32 + rt * 16;

  int m     = lane & 15;
  int khalf = lane >> 4;

  f32x8 acc0 = {0.f,0.f,0.f,0.f,0.f,0.f,0.f,0.f};
  f32x8 acc1 = acc0, acc2 = acc0, acc3 = acc0;

  int arow = rowbase + m;
  if (arow >= Nt) arow = Nt - 1;           // clamp: WMMA needs EXEC all-ones
  const unsigned short* arowp = hbf + (size_t)arow * 256 + khalf * 16;

  union { uint4 q[2]; bf16x16 v; } fa, fb;

  #pragma unroll
  for (int kt = 0; kt < 8; ++kt) {
    // A fragment: lane m holds K = kt*32 + khalf*16 + {0..15}
    const unsigned short* ap = arowp + kt * 32;
    fa.q[0] = ((const uint4*)ap)[0];
    fa.q[1] = ((const uint4*)(ap + 8))[0];

    const unsigned short* bp = wpack + ((size_t)(kt * 16 + cg * 4) * 32 + lane) * 16;
    fb.q[0] = ((const uint4*)bp)[0];
    fb.q[1] = ((const uint4*)bp)[1];
    acc0 = __builtin_amdgcn_wmma_f32_16x16x32_bf16(false, fa.v, false, fb.v,
                                                   (short)0, acc0, false, false);
    fb.q[0] = ((const uint4*)(bp + 512))[0];
    fb.q[1] = ((const uint4*)(bp + 512))[1];
    acc1 = __builtin_amdgcn_wmma_f32_16x16x32_bf16(false, fa.v, false, fb.v,
                                                   (short)0, acc1, false, false);
    fb.q[0] = ((const uint4*)(bp + 1024))[0];
    fb.q[1] = ((const uint4*)(bp + 1024))[1];
    acc2 = __builtin_amdgcn_wmma_f32_16x16x32_bf16(false, fa.v, false, fb.v,
                                                   (short)0, acc2, false, false);
    fb.q[0] = ((const uint4*)(bp + 1536))[0];
    fb.q[1] = ((const uint4*)(bp + 1536))[1];
    acc3 = __builtin_amdgcn_wmma_f32_16x16x32_bf16(false, fa.v, false, fb.v,
                                                   (short)0, acc3, false, false);
  }

  // C/D layout: VGPR r, lanes 0-15 -> M=r, lanes 16-31 -> M=r+8; N = lane&15
  int ncol0 = cg * 64 + (lane & 15);
  int mbase = (lane >> 4) * 8;
  #pragma unroll
  for (int r = 0; r < 8; ++r) {
    int row = rowbase + mbase + r;
    if (row < Nt) {
      float* op = xs + (size_t)row * 256 + ncol0;
      op[0]  = acc0[r];
      op[16] = acc1[r];
      op[32] = acc2[r];
      op[48] = acc3[r];
    }
  }
}

// Per-node attention logits: al_s/al_d [Nt,H]
__global__ void attn_node(const float* __restrict__ xs,
                          const float* __restrict__ a_src,
                          const float* __restrict__ a_dst,
                          float* __restrict__ al_s,
                          float* __restrict__ al_d, int Nt) {
  int t = blockIdx.x * blockDim.x + threadIdx.x;
  if (t >= Nt * NHEADS) return;
  int n = t >> 2, hh = t & 3;
  const float* xp = xs + (size_t)n * 256 + hh * 64;
  const float* as = a_src + hh * 64;
  const float* ad = a_dst + hh * 64;
  float ss = 0.f, sd = 0.f;
  #pragma unroll 8
  for (int c = 0; c < 64; ++c) { float v = xp[c]; ss += v * as[c]; sd += v * ad[c]; }
  al_s[t] = ss;
  al_d[t] = sd;
}

__device__ __forceinline__ void decode_edge(int e, const int* __restrict__ ei,
                                            const float* __restrict__ eattr,
                                            const float* __restrict__ lattr,
                                            int& src, int& dst,
                                            float& a0, float& a1, float& a2) {
  if (e < NEDGES) {
    src = ei[e]; dst = ei[NEDGES + e];
    a0 = eattr[e * 3]; a1 = eattr[e * 3 + 1]; a2 = eattr[e * 3 + 2];
  } else if (e < NEDGES + NNODES) {
    src = NNODES; dst = e - NEDGES; a0 = 0.5f; a1 = 0.f; a2 = 0.f;
  } else if (e < NEDGES + 2 * NNODES) {
    src = e - NEDGES - NNODES; dst = NNODES; a0 = 0.5f; a1 = 0.f; a2 = 0.f;
  } else {
    int n = e - NEDGES - 2 * NNODES;
    src = n; dst = n;
    a0 = lattr[n * 3]; a1 = lattr[n * 3 + 1]; a2 = lattr[n * 3 + 2];
  }
}

// Pass 1: alpha = leaky_relu(al_s[src]+al_d[dst]+al_e); segment max via
// ordered-uint atomicMax (sentinel 0 == -inf; every dst has a self-loop).
__global__ void edge_alpha(const int* __restrict__ ei,
                           const float* __restrict__ eattr,
                           const float* __restrict__ lattr,
                           const float* __restrict__ al_s,
                           const float* __restrict__ al_d,
                           const float* __restrict__ g,
                           float* __restrict__ abuf,
                           unsigned int* __restrict__ mx, int Et) {
  int t = blockIdx.x * blockDim.x + threadIdx.x;
  if (t >= Et * NHEADS) return;
  int e = t >> 2, hh = t & 3;
  int src, dst; float a0, a1, a2;
  decode_edge(e, ei, eattr, lattr, src, dst, a0, a1, a2);
  float ale = a0 * g[0 * 4 + hh] + a1 * g[1 * 4 + hh] + a2 * g[2 * 4 + hh];
  float alpha = al_s[src * 4 + hh] + al_d[dst * 4 + hh] + ale;
  alpha = (alpha > 0.f) ? alpha : 0.2f * alpha;
  abuf[t] = alpha;
  unsigned u = __float_as_uint(alpha);
  u = (u & 0x80000000u) ? ~u : (u | 0x80000000u);
  atomicMax(&mx[dst * 4 + hh], u);
}

// Pass 2: a = exp(alpha - mx[dst]); segment sum
__global__ void edge_exp(const int* __restrict__ ei,
                         const float* __restrict__ eattr,
                         const float* __restrict__ lattr,
                         const unsigned int* __restrict__ mx,
                         float* __restrict__ abuf,
                         float* __restrict__ ssum, int Et) {
  int t = blockIdx.x * blockDim.x + threadIdx.x;
  if (t >= Et * NHEADS) return;
  int e = t >> 2, hh = t & 3;
  int src, dst; float a0, a1, a2;
  decode_edge(e, ei, eattr, lattr, src, dst, a0, a1, a2);
  unsigned u = mx[dst * 4 + hh];
  float mxv = (u == 0u) ? -3.0e38f
            : ((u & 0x80000000u) ? __uint_as_float(u ^ 0x80000000u)
                                 : __uint_as_float(~u));
  float a = expf(abuf[t] - mxv);
  abuf[t] = a;
  atomicAdd(&ssum[dst * 4 + hh], a);
}

// Pass 3: scatter messages. Thread = (edge, 32-channel slab); head fixed/slab.
__global__ void edge_msg(const int* __restrict__ ei,
                         const float* __restrict__ eattr,
                         const float* __restrict__ lattr,
                         const float* __restrict__ abuf,
                         const float* __restrict__ ssum,
                         const float* __restrict__ xs,
                         float* __restrict__ outagg, int Et) {
  int t = blockIdx.x * blockDim.x + threadIdx.x;
  if (t >= Et * 8) return;
  int e = t >> 3, q = t & 7;
  int hh = q >> 1;                       // 32-channel slab lies in one head
  int src, dst; float a0, a1, a2;
  decode_edge(e, ei, eattr, lattr, src, dst, a0, a1, a2);
  float w = abuf[e * 4 + hh] / (ssum[dst * 4 + hh] + 1e-16f);
  const float* xp = xs + (size_t)src * 256 + q * 32;
  float* op = outagg + (size_t)dst * 256 + q * 32;
  #pragma unroll
  for (int c = 0; c < 32; ++c) atomicAdd(&op[c], xp[c] * w);
}

// h = relu(LN(out + bias) + h); refresh bf16 mirror; snapshot virtual row
__global__ void node_update(const float* __restrict__ outagg,
                            const float* __restrict__ bias,
                            const float* __restrict__ lng,
                            const float* __restrict__ lnb,
                            float* __restrict__ h,
                            unsigned short* __restrict__ hbf,
                            float* __restrict__ save_row, int Nt) {
  __shared__ float sm[256];
  int n = blockIdx.x;
  int c = threadIdx.x;
  float v = outagg[(size_t)n * 256 + c] + bias[c];
  float m = block_sum_256(v, sm) * (1.f / 256.f);
  float d = v - m;
  float var = block_sum_256(d * d, sm) * (1.f / 256.f);
  float hn = d * rsqrtf(var + LN_EPS) * lng[c] + lnb[c];
  float r = fmaxf(hn + h[(size_t)n * 256 + c], 0.f);
  h[(size_t)n * 256 + c] = r;
  hbf[(size_t)n * 256 + c] = f2bf(r);
  if (n == Nt - 1) save_row[c] = r;
}

// Agent encoder (single block)
__global__ void agent_encode(const float* __restrict__ af,
                             const float* __restrict__ ag_w,
                             const float* __restrict__ ag_b,
                             const float* __restrict__ ag_g,
                             const float* __restrict__ ag_bt,
                             float* __restrict__ agent_vec) {
  __shared__ float sm[256];
  __shared__ float xr[10];
  int c = threadIdx.x;
  if (c < 10) xr[c] = af[c];
  __syncthreads();
  float acc = ag_b[c];
  #pragma unroll
  for (int k = 0; k < 10; ++k) acc += xr[k] * ag_w[k * 256 + c];
  float m = block_sum_256(acc, sm) * (1.f / 256.f);
  float d = acc - m;
  float var = block_sum_256(d * d, sm) * (1.f / 256.f);
  float v = d * rsqrtf(var + LN_EPS) * ag_g[c] + ag_bt[c];
  agent_vec[c] = fmaxf(v, 0.f);
}

// Dueling heads (single block): comb[1280] = [jk row N (1024), agent (256)]
__global__ void heads_kernel(const float* __restrict__ save_rows,
                             const float* __restrict__ agent_vec,
                             const float* __restrict__ v_w1, const float* __restrict__ v_b1,
                             const float* __restrict__ v_g,  const float* __restrict__ v_bt,
                             const float* __restrict__ v_w2, const float* __restrict__ v_b2,
                             const float* __restrict__ a_w1, const float* __restrict__ a_b1,
                             const float* __restrict__ a_g,  const float* __restrict__ a_bt,
                             const float* __restrict__ a_w2, const float* __restrict__ a_b2,
                             float* __restrict__ out) {
  __shared__ float comb[1280];
  __shared__ float sm[256];
  __shared__ float hv[256];
  __shared__ float ha[256];
  __shared__ float adv[9];
  int tid = threadIdx.x;
  for (int i = tid; i < 1280; i += 256)
    comb[i] = (i < 1024) ? save_rows[i] : agent_vec[i - 1024];
  __syncthreads();

  float acc = v_b1[tid];
  for (int k = 0; k < 1280; ++k) acc += comb[k] * v_w1[k * 256 + tid];
  float m = block_sum_256(acc, sm) * (1.f / 256.f);
  float d = acc - m;
  float var = block_sum_256(d * d, sm) * (1.f / 256.f);
  hv[tid] = fmaxf(d * rsqrtf(var + LN_EPS) * v_g[tid] + v_bt[tid], 0.f);

  acc = a_b1[tid];
  for (int k = 0; k < 1280; ++k) acc += comb[k] * a_w1[k * 256 + tid];
  m = block_sum_256(acc, sm) * (1.f / 256.f);
  d = acc - m;
  var = block_sum_256(d * d, sm) * (1.f / 256.f);
  ha[tid] = fmaxf(d * rsqrtf(var + LN_EPS) * a_g[tid] + a_bt[tid], 0.f);
  __syncthreads();

  float val = block_sum_256(hv[tid] * v_w2[tid], sm) + v_b2[0];

  if (tid < 9) {
    float s = a_b2[tid];
    for (int c = 0; c < 256; ++c) s += ha[c] * a_w2[c * 9 + tid];
    adv[tid] = s;
  }
  __syncthreads();
  if (tid == 0) {
    float mean = 0.f;
    for (int j = 0; j < 9; ++j) mean += adv[j];
    mean *= (1.f / 9.f);
    for (int j = 0; j < 9; ++j) out[j] = val + adv[j] - mean;
  }
}

// ---------------------------------------------------------------------------
extern "C" void kernel_launch(void* const* d_in, const int* in_sizes, int n_in,
                              void* d_out, int out_size, void* d_ws, size_t ws_size,
                              hipStream_t stream) {
  (void)in_sizes; (void)n_in; (void)out_size; (void)ws_size;
  const float* x      = (const float*)d_in[0];
  const int*   ei     = (const int*)  d_in[1];
  const float* eattr  = (const float*)d_in[2];
  const float* agf    = (const float*)d_in[3];
  const float* enc_w  = (const float*)d_in[4];
  const float* enc_b  = (const float*)d_in[5];
  const float* enc_g  = (const float*)d_in[6];
  const float* enc_bt = (const float*)d_in[7];
  const float* vn_base= (const float*)d_in[8];
  const float* vn_w1  = (const float*)d_in[9];
  const float* vn_b1  = (const float*)d_in[10];
  const float* vn_g   = (const float*)d_in[11];
  const float* vn_bt  = (const float*)d_in[12];
  const float* vn_w2  = (const float*)d_in[13];
  const float* vn_b2  = (const float*)d_in[14];
  const float* gat_w  = (const float*)d_in[15];
  const float* att_src= (const float*)d_in[16];
  const float* att_dst= (const float*)d_in[17];
  const float* att_edge=(const float*)d_in[18];
  const float* edge_w = (const float*)d_in[19];
  const float* gat_b  = (const float*)d_in[20];
  const float* ln_g   = (const float*)d_in[21];
  const float* ln_b   = (const float*)d_in[22];
  const float* ag_w   = (const float*)d_in[23];
  const float* ag_b   = (const float*)d_in[24];
  const float* ag_g   = (const float*)d_in[25];
  const float* ag_bt  = (const float*)d_in[26];
  const float* v_w1   = (const float*)d_in[27];
  const float* v_b1   = (const float*)d_in[28];
  const float* v_g    = (const float*)d_in[29];
  const float* v_bt   = (const float*)d_in[30];
  const float* v_w2   = (const float*)d_in[31];
  const float* v_b2   = (const float*)d_in[32];
  const float* a_w1   = (const float*)d_in[33];
  const float* a_b1   = (const float*)d_in[34];
  const float* a_g    = (const float*)d_in[35];
  const float* a_bt   = (const float*)d_in[36];
  const float* a_w2   = (const float*)d_in[37];
  const float* a_b2   = (const float*)d_in[38];

  const int Nt = NNODES + 1;
  const int Et = NEDGES + 2 * NNODES + Nt;

  size_t off = 0;
  char* base = (char*)d_ws;
  auto alloc = [&](size_t bytes) -> void* {
    void* p = base + off;
    off = (off + bytes + 255) & ~(size_t)255;
    return p;
  };
  float*          h      = (float*)         alloc((size_t)Nt * 256 * 4);
  unsigned short* hbf    = (unsigned short*)alloc((size_t)Nt * 256 * 2);
  float*          xs     = (float*)         alloc((size_t)Nt * 256 * 4);
  float*          outagg = (float*)         alloc((size_t)Nt * 256 * 4);
  float*          ctx    = (float*)         alloc(256 * 4);
  float*          deg    = (float*)         alloc((size_t)Nt * 4);
  float*          asum   = (float*)         alloc((size_t)Nt * 3 * 4);
  float*          lattr  = (float*)         alloc((size_t)Nt * 3 * 4);
  float*          al_s   = (float*)         alloc((size_t)Nt * 4 * 4);
  float*          al_d   = (float*)         alloc((size_t)Nt * 4 * 4);
  unsigned int*   mx     = (unsigned int*)  alloc((size_t)Nt * 4 * 4);
  float*          ssum   = (float*)         alloc((size_t)Nt * 4 * 4);
  float*          abuf   = (float*)         alloc((size_t)Et * 4 * 4);
  unsigned short* wpack  = (unsigned short*)alloc((size_t)NLAYERS * 65536 * 2);
  float*          gbuf   = (float*)         alloc(NLAYERS * 3 * 4 * 4);
  float*          save_rows = (float*)      alloc(4 * 256 * 4);
  float*          agent_vec = (float*)      alloc(256 * 4);

  hipMemsetAsync(ctx, 0, 256 * 4, stream);
  hipMemsetAsync(deg, 0, (size_t)Nt * 4, stream);
  hipMemsetAsync(asum, 0, (size_t)Nt * 3 * 4, stream);

  pack_gat_weights<<<(NLAYERS * 65536 + 255) / 256, 256, 0, stream>>>(gat_w, wpack);
  compute_edge_gain<<<1, 64, 0, stream>>>(edge_w, att_edge, gbuf);
  encode_nodes<<<NNODES, 256, 0, stream>>>(x, enc_w, enc_b, enc_g, enc_bt, h, hbf, ctx);
  virtual_node<<<1, 256, 0, stream>>>(ctx, vn_base, vn_w1, vn_b1, vn_g, vn_bt,
                                      vn_w2, vn_b2, h, hbf, save_rows);
  edge_degree<<<(NEDGES + 2 * NNODES + 255) / 256, 256, 0, stream>>>(ei, eattr, deg, asum);
  finalize_loop_attr<<<(Nt + 255) / 256, 256, 0, stream>>>(deg, asum, lattr);

  for (int l = 0; l < NLAYERS; ++l) {
    hipMemsetAsync(outagg, 0, (size_t)Nt * 256 * 4, stream);
    hipMemsetAsync(ssum, 0, (size_t)Nt * 4 * 4, stream);
    hipMemsetAsync(mx, 0, (size_t)Nt * 4 * 4, stream);   // 0 == -inf sentinel

    gemm_xs_wmma<<<(Nt + 31) / 32, 256, 0, stream>>>(hbf, wpack + (size_t)l * 65536, xs, Nt);
    attn_node<<<(Nt * 4 + 255) / 256, 256, 0, stream>>>(xs, att_src + l * 256,
                                                        att_dst + l * 256, al_s, al_d, Nt);
    edge_alpha<<<(Et * 4 + 255) / 256, 256, 0, stream>>>(ei, eattr, lattr, al_s, al_d,
                                                         gbuf + l * 12, abuf, mx, Et);
    edge_exp<<<(Et * 4 + 255) / 256, 256, 0, stream>>>(ei, eattr, lattr, mx, abuf, ssum, Et);
    edge_msg<<<(Et * 8 + 255) / 256, 256, 0, stream>>>(ei, eattr, lattr, abuf, ssum,
                                                       xs, outagg, Et);
    node_update<<<Nt, 256, 0, stream>>>(outagg, gat_b + l * 256, ln_g + l * 256,
                                        ln_b + l * 256, h, hbf,
                                        save_rows + (size_t)(l + 1) * 256, Nt);
  }

  agent_encode<<<1, 256, 0, stream>>>(agf, ag_w, ag_b, ag_g, ag_bt, agent_vec);
  heads_kernel<<<1, 256, 0, stream>>>(save_rows, agent_vec,
                                      v_w1, v_b1, v_g, v_bt, v_w2, v_b2,
                                      a_w1, a_b1, a_g, a_bt, a_w2, a_b2,
                                      (float*)d_out);
}
